// MultiheadAttention_13606456394410
// MI455X (gfx1250) — matrix operand order repaired and checked by reading, not compiled, not run on hardware
//
#include <hip/hip_runtime.h>

typedef __attribute__((ext_vector_type(16))) __bf16 v16bf;
typedef __attribute__((ext_vector_type(8)))  float  v8f;
typedef __attribute__((ext_vector_type(4)))  int    v4i;

#define NSEQ    2304     // 48*48
#define DMODEL  512
#define NHEADS  8
#define DPHEAD  64
#define MROWS   4608     // 2 * NSEQ
#define WDIM    48

#define GLOBAL_AS __attribute__((address_space(1)))
#define LDS_AS    __attribute__((address_space(3)))

#if __has_builtin(__builtin_amdgcn_global_load_async_to_lds_b128)
#define HAVE_ASYNC_LDS 1
#else
#define HAVE_ASYNC_LDS 0
#endif

static __device__ __forceinline__ v8f wmma_bf16(v16bf a, v16bf b, v8f c) {
  // D = A(16x32 bf16) * B(32x16 bf16) + C(16x16 f32)
  return __builtin_amdgcn_wmma_f32_16x16x32_bf16(false, a, false, b, (short)0, c, false, false);
}

// A-fragment slot mapping: identity (contiguous K order, matches B)
static __device__ __forceinline__ int aslot(int c) {
  return c;
}

// 16-byte global->LDS block copy; async on CDNA5 when the builtin exists.
static __device__ __forceinline__ void copy16(const __bf16* g, __bf16* l) {
#if HAVE_ASYNC_LDS
  __builtin_amdgcn_global_load_async_to_lds_b128((GLOBAL_AS v4i*)g, (LDS_AS v4i*)l, 0, 0);
#else
  *(uint4*)l = *(const uint4*)g;
#endif
}

static __device__ __forceinline__ void wait_async() {
#if HAVE_ASYNC_LDS
#if __has_builtin(__builtin_amdgcn_s_wait_asynccnt)
  __builtin_amdgcn_s_wait_asynccnt(0);
#else
  asm volatile("s_wait_asynccnt 0x0" ::: "memory");
#endif
#endif
}

// ---------------------------------------------------------------------------
// Kernel 1: QKV projection, 64x64 tile / block (8 waves).  LDS tiles stored in
// fragment order: A natural [m][k], B transposed [n][k].  Q,K raw f32 out
// (RoPE later) in [b][h][n][dv]; V written bf16 pre-transposed [b][h][dv][n].
// ---------------------------------------------------------------------------
__global__ __launch_bounds__(256)
void qkv_gemm_kernel(const float* __restrict__ x,
                     const float* __restrict__ wq,
                     const float* __restrict__ wk,
                     const float* __restrict__ wv,
                     float* __restrict__ qraw,
                     float* __restrict__ kraw,
                     __bf16* __restrict__ vbT)
{
  __shared__ __align__(64) __bf16 As2[64][48];   // 64 x 32 used, slot order
  __shared__ __align__(64) __bf16 Bst[64][48];   // [n][k] 64 x 32 used
  const int tid  = threadIdx.x;
  const int lane = tid & 31;
  const int wave = tid >> 5;
  const int l  = lane & 15;
  const int g  = lane >> 4;
  const int mw = wave >> 1;       // 16-row strip
  const int nw = wave & 1;        // 32-col half
  const int m0 = blockIdx.x * 64;
  const int h  = blockIdx.y;
  const int mat = blockIdx.z;     // 0:q 1:k 2:v
  const float* w = (mat == 0) ? wq : (mat == 1) ? wk : wv;

  v8f c0 = {}; v8f c1 = {};

  for (int k0 = 0; k0 < DMODEL; k0 += 32) {
    #pragma unroll
    for (int r = 0; r < 8; ++r) {                    // A: coalesced, slot store
      int e = tid + r * 256;
      int row = e >> 5, col = e & 31;
      As2[row][aslot(col)] = (__bf16)x[(long)(m0 + row) * DMODEL + k0 + col];
    }
    #pragma unroll
    for (int r = 0; r < 8; ++r) {                    // B: W[d][v][h] -> [v][d]
      int e = tid + r * 256;
      int krow = e >> 6, col = e & 63;
      Bst[col][krow] = (__bf16)w[(long)(k0 + krow) * (DPHEAD * NHEADS) + col * NHEADS + h];
    }
    __syncthreads();

    v16bf af = *(const v16bf*)&As2[mw * 16 + l][g * 16];
    v16bf b0 = *(const v16bf*)&Bst[nw * 32 + l][g * 16];
    v16bf b1 = *(const v16bf*)&Bst[nw * 32 + 16 + l][g * 16];
    c0 = wmma_bf16(af, b0, c0);
    c1 = wmma_bf16(af, b1, c1);
    __syncthreads();
  }

  #pragma unroll
  for (int r = 0; r < 8; ++r) {
    int m  = m0 + mw * 16 + r + 8 * g;
    int b  = m / NSEQ;
    int nn = m - b * NSEQ;
    long bh  = (long)b * NHEADS + h;
    int v0c = nw * 32 + l;
    if (mat == 0) {
      long base = (bh * NSEQ + nn) * DPHEAD;
      qraw[base + v0c] = c0[r]; qraw[base + v0c + 16] = c1[r];
    } else if (mat == 1) {
      long base = (bh * NSEQ + nn) * DPHEAD;
      kraw[base + v0c] = c0[r]; kraw[base + v0c + 16] = c1[r];
    } else {                                   // V transposed: [b][h][dv][n]
      long tbase = bh * DPHEAD;
      vbT[(tbase + v0c) * NSEQ + nn]      = (__bf16)c0[r];
      vbT[(tbase + v0c + 16) * NSEQ + nn] = (__bf16)c1[r];
    }
  }
}

// ---------------------------------------------------------------------------
// Kernel 2: 2-D RoPE on Q,K (interleaved pairs); folds 1/sqrt(64) into Q.
// ---------------------------------------------------------------------------
__global__ __launch_bounds__(256)
void rope_kernel(const float* __restrict__ qraw,
                 const float* __restrict__ kraw,
                 const float* __restrict__ angles,   // [2][32]
                 __bf16* __restrict__ qb,
                 __bf16* __restrict__ kb)
{
  int idx  = blockIdx.x * 256 + threadIdx.x;  // over 2*8*2304*32
  int c    = idx & 31;
  int rest = idx >> 5;
  int nn   = rest % NSEQ;
  int bh   = rest / NSEQ;
  int i    = nn / WDIM;
  int j    = nn - i * WDIM;
  float ang = (float)i * angles[c] + (float)j * angles[32 + c];
  float sn, cs;
  __sincosf(ang, &sn, &cs);
  long base = ((long)bh * NSEQ + nn) * DPHEAD + 2 * c;
  float q0 = qraw[base], q1 = qraw[base + 1];
  qb[base]     = (__bf16)((q0 * cs - q1 * sn) * 0.125f);
  qb[base + 1] = (__bf16)((q0 * sn + q1 * cs) * 0.125f);
  float k0 = kraw[base], k1 = kraw[base + 1];
  kb[base]     = (__bf16)(k0 * cs - k1 * sn);
  kb[base + 1] = (__bf16)(k0 * sn + k1 * cs);
}

// ---------------------------------------------------------------------------
// Kernel 3: flash attention. 64 q-rows x one (b,h) per block; 32-key tiles.
// All staging is 16-byte block copies (async-to-LDS on CDNA5).  Fragment
// reads are contiguous v16bf LDS loads.
// ---------------------------------------------------------------------------
__global__ __launch_bounds__(256)
void attn_kernel(const __bf16* __restrict__ qb,
                 const __bf16* __restrict__ kb,
                 const __bf16* __restrict__ vbT,     // [b][h][dv][n]
                 float* __restrict__ ob)
{
  __shared__ __align__(64) __bf16 Qs2[64][80];      // 64 x 64 used, slot order
  __shared__ __align__(64) __bf16 Ks[32][80];       // [key][dv], natural
  __shared__ __align__(64) __bf16 Vst[64][48];      // [dv][key]
  __shared__ __align__(64) __bf16 Ps2[4][16][48];   // per-strip P, slot order

  const int tid  = threadIdx.x;
  const int lane = tid & 31;
  const int wave = tid >> 5;
  const int l  = lane & 15;
  const int g  = lane >> 4;
  const int mw = wave >> 1;
  const int nw = wave & 1;
  const int q0row    = blockIdx.x * 64;
  const long seqbase = (long)blockIdx.y * NSEQ;
  const long vbase   = (long)blockIdx.y * DPHEAD * NSEQ;

  // stage Q (8 blocks of 16B per row); natural (contiguous K) layout
  #pragma unroll
  for (int it = 0; it < 2; ++it) {
    int e = tid + it * 256;
    int row = e >> 3, bb = e & 7;
    int sb = bb;
    copy16(&qb[(seqbase + q0row + row) * DPHEAD + sb * 8], &Qs2[row][bb * 8]);
  }
  wait_async();
  __syncthreads();

  v16bf aq0 = *(const v16bf*)&Qs2[mw * 16 + l][g * 16];
  v16bf aq1 = *(const v16bf*)&Qs2[mw * 16 + l][32 + g * 16];

  v8f o0 = {}, o1 = {};
  float mstate[8], ssum[8];
  #pragma unroll
  for (int r = 0; r < 8; ++r) { mstate[r] = -3.0e38f; ssum[r] = 0.f; }

  for (int kt = 0; kt < NSEQ; kt += 32) {
    {                                             // K tile: 32 rows x 8 blocks
      int row = tid >> 3, seg = tid & 7;
      copy16(&kb[(seqbase + kt + row) * DPHEAD + seg * 8], &Ks[row][seg * 8]);
    }
    {                                             // V^T tile: 64 rows x 4 blocks
      int dv = tid >> 2, seg = tid & 3;
      copy16(&vbT[vbase + (long)dv * NSEQ + kt + seg * 8], &Vst[dv][seg * 8]);
    }
    wait_async();
    __syncthreads();

    // S(16q x 32keys) = Q * K^T
    v16bf bk00 = *(const v16bf*)&Ks[l][16 * g];
    v16bf bk01 = *(const v16bf*)&Ks[l][32 + 16 * g];
    v16bf bk10 = *(const v16bf*)&Ks[16 + l][16 * g];
    v16bf bk11 = *(const v16bf*)&Ks[16 + l][32 + 16 * g];
    v8f s0 = {}, s1 = {};
    s0 = wmma_bf16(aq0, bk00, s0);
    s0 = wmma_bf16(aq1, bk01, s0);
    s1 = wmma_bf16(aq0, bk10, s1);
    s1 = wmma_bf16(aq1, bk11, s1);

    // online softmax: lane holds rows r+8g at key column l (s0) / l+16 (s1)
    #pragma unroll
    for (int r = 0; r < 8; ++r) {
      float tmax = fmaxf(s0[r], s1[r]);
      #pragma unroll
      for (int off = 1; off < 16; off <<= 1)
        tmax = fmaxf(tmax, __shfl_xor(tmax, off, 32));
      float mnew  = fmaxf(mstate[r], tmax);
      float alpha = __expf(mstate[r] - mnew);
      mstate[r] = mnew;
      float p0 = __expf(s0[r] - mnew);
      float p1 = __expf(s1[r] - mnew);
      float rs = p0 + p1;
      #pragma unroll
      for (int off = 1; off < 16; off <<= 1)
        rs += __shfl_xor(rs, off, 32);
      ssum[r] = ssum[r] * alpha + rs;
      o0[r] *= alpha;
      o1[r] *= alpha;
      if (nw == 0) {                              // P -> A-slot layout
        Ps2[mw][r + 8 * g][aslot(l)]      = (__bf16)p0;
        Ps2[mw][r + 8 * g][aslot(l + 16)] = (__bf16)p1;
      }
    }
    __syncthreads();

    // O(16q x 32dv) += P(16x32) * V(32keys x dv)
    v16bf ap  = *(const v16bf*)&Ps2[mw][l][g * 16];
    v16bf bv0 = *(const v16bf*)&Vst[nw * 32 + l][16 * g];
    v16bf bv1 = *(const v16bf*)&Vst[nw * 32 + 16 + l][16 * g];
    o0 = wmma_bf16(ap, bv0, o0);
    o1 = wmma_bf16(ap, bv1, o1);
    __syncthreads();
  }

  #pragma unroll
  for (int r = 0; r < 8; ++r) {
    float inv = 1.0f / ssum[r];
    int row = q0row + mw * 16 + r + 8 * g;
    long base = (seqbase + row) * DPHEAD;
    ob[base + nw * 32 + l]      = o0[r] * inv;
    ob[base + nw * 32 + 16 + l] = o1[r] * inv;
  }
}

// ---------------------------------------------------------------------------
// Kernel 4: output projection.  out[m,d] = sum_{k=(h,v)} ob[b,h,nn,v]*wo[d,v,h]
// ---------------------------------------------------------------------------
__global__ __launch_bounds__(256)
void oproj_kernel(const float* __restrict__ ob,
                  const float* __restrict__ wo,
                  float* __restrict__ out)
{
  __shared__ __align__(64) __bf16 As2[64][48];
  __shared__ __align__(64) __bf16 Bst[64][48];
  const int tid  = threadIdx.x;
  const int lane = tid & 31;
  const int wave = tid >> 5;
  const int l  = lane & 15;
  const int g  = lane >> 4;
  const int mw = wave >> 1;
  const int nw = wave & 1;
  const int m0 = blockIdx.x * 64;
  const int n0 = blockIdx.y * 64;

  v8f c0 = {}, c1 = {};

  for (int k0 = 0; k0 < DMODEL; k0 += 32) {
    #pragma unroll
    for (int r = 0; r < 8; ++r) {
      int e = tid + r * 256;
      int row = e >> 5, col = e & 31;
      int m = m0 + row;
      int k = k0 + col;
      int b = m / NSEQ, nn = m - b * NSEQ;
      int h = k >> 6, v = k & 63;
      As2[row][aslot(col)] =
          (__bf16)ob[(((long)b * NHEADS + h) * NSEQ + nn) * DPHEAD + v];
    }
    #pragma unroll
    for (int r = 0; r < 8; ++r) {
      int e = tid + r * 256;
      int col = e >> 5, krow = e & 31;
      int k = k0 + krow;
      int h = k >> 6, v = k & 63;
      Bst[col][krow] = (__bf16)wo[(long)(n0 + col) * DMODEL + v * NHEADS + h];
    }
    __syncthreads();

    v16bf af = *(const v16bf*)&As2[mw * 16 + l][g * 16];
    v16bf b0 = *(const v16bf*)&Bst[nw * 32 + l][g * 16];
    v16bf b1 = *(const v16bf*)&Bst[nw * 32 + 16 + l][g * 16];
    c0 = wmma_bf16(af, b0, c0);
    c1 = wmma_bf16(af, b1, c1);
    __syncthreads();
  }

  #pragma unroll
  for (int r = 0; r < 8; ++r) {
    int m = m0 + mw * 16 + r + 8 * g;
    out[(long)m * DMODEL + n0 + nw * 32 + l]      = c0[r];
    out[(long)m * DMODEL + n0 + nw * 32 + 16 + l] = c1[r];
  }
}

// ---------------------------------------------------------------------------
extern "C" void kernel_launch(void* const* d_in, const int* in_sizes, int n_in,
                              void* d_out, int out_size, void* d_ws, size_t ws_size,
                              hipStream_t stream) {
  (void)in_sizes; (void)n_in; (void)out_size; (void)ws_size;
  const float* x      = (const float*)d_in[0];
  const float* wq     = (const float*)d_in[1];
  const float* wk     = (const float*)d_in[2];
  const float* wv     = (const float*)d_in[3];
  const float* wo     = (const float*)d_in[4];
  const float* angles = (const float*)d_in[5];

  const long per = (long)2 * NHEADS * NSEQ * DPHEAD;   // 2,359,296 elems
  float*  qraw = (float*)d_ws;
  float*  kraw = qraw + per;
  float*  ob   = kraw + per;
  __bf16* qb   = (__bf16*)(ob + per);
  __bf16* kb   = qb + per;
  __bf16* vbT  = kb + per;

  qkv_gemm_kernel<<<dim3(MROWS / 64, NHEADS, 3), 256, 0, stream>>>(
      x, wq, wk, wv, qraw, kraw, vbT);
  rope_kernel<<<(2 * NHEADS * NSEQ * 32) / 256, 256, 0, stream>>>(
      qraw, kraw, angles, qb, kb);
  attn_kernel<<<dim3(NSEQ / 64, 2 * NHEADS), 256, 0, stream>>>(
      qb, kb, vbT, ob);
  oproj_kernel<<<dim3(MROWS / 64, DMODEL / 64), 256, 0, stream>>>(
      ob, wo, (float*)d_out);
}
